// DeepClustering_18571438588712
// MI455X (gfx1250) — matrix operand order repaired and checked by reading, not compiled, not run on hardware
//
#include <hip/hip_runtime.h>
#include <hip/hip_bf16.h>

// ---------------------------------------------------------------------------
// Problem constants
// ---------------------------------------------------------------------------
#define BATCH   64
#define SEQ     128
#define D_IN    16
#define D_MODEL 256
#define NHEAD   8
#define DHEAD   32
#define D_FF    1024
#define NROWS   8192      // B*S
#define KNNS    10

// ---------------------------------------------------------------------------
// CDNA5 async global->LDS copy (gated: safe on toolchains without builtins)
// Builtin signature (from compiler diagnostic): params are pointers to v4i.
// ---------------------------------------------------------------------------
#define AS1 __attribute__((address_space(1)))
#define AS3 __attribute__((address_space(3)))
typedef __attribute__((ext_vector_type(4))) int v4i;
#if defined(__AMDGCN__) && \
    __has_builtin(__builtin_amdgcn_global_load_async_to_lds_b128) && \
    __has_builtin(__builtin_amdgcn_s_wait_asynccnt)
#define HAVE_ASYNC_LDS 1
#define GLOAD_ASYNC_B128(src, dst) \
    __builtin_amdgcn_global_load_async_to_lds_b128((AS1 v4i*)(src), (AS3 v4i*)(dst), 0, 0)
#else
#define HAVE_ASYNC_LDS 0
#endif

// ---------------------------------------------------------------------------
// WMMA types (CDNA5 / gfx1250, wave32)
// ---------------------------------------------------------------------------
typedef __bf16 bf16_t;
typedef __attribute__((ext_vector_type(4)))  __bf16 v4bf;
typedef __attribute__((ext_vector_type(8)))  __bf16 v8bf;
typedef __attribute__((ext_vector_type(16))) __bf16 v16bf;
typedef __attribute__((ext_vector_type(8)))  float  v8f;

#define TILE_M 32
#define TILE_N 128
#define TILE_K 32
#define LDSS   40          // LDS row stride in bf16 units (80 B, 16B-multiple)

// ---------------------------------------------------------------------------
// Fast GEMM:  C[M,N] = act(A[M,K] @ B[K,N] + bias[N])
// REQUIRES M%32==0, N%128==0, K%32==0  ->  zero guards, zero divergence.
// 8 waves tiled 2(M) x 4(N); each wave owns TWO 16x16 tiles sharing one A
// fragment -> 2x v_wmma_f32_16x16x32_bf16 per k-step. Register double-
// buffered staging; B transposed in registers (4x4 sub-blocks) -> b64 stores.
// ---------------------------------------------------------------------------
__global__ __launch_bounds__(256)
void gemm_wmma_kernel(const float* __restrict__ A, const float* __restrict__ B,
                      const float* __restrict__ bias, float* __restrict__ C,
                      int M, int N, int K, int relu)
{
    __shared__ __align__(16) bf16_t As[TILE_M * LDSS];     // [m][k]
    __shared__ __align__(16) bf16_t BsT[TILE_N * LDSS];    // [n][k] (transposed)

    const int tid   = threadIdx.x;
    const int wave  = tid >> 5;
    const int lane  = tid & 31;
    const int waveM = wave >> 2;       // 0..1
    const int waveN = wave & 3;        // 0..3
    const int hl    = lane >> 4;       // K-group select
    const int l16   = lane & 15;

    const int blockM = blockIdx.x * TILE_M;
    const int blockN = blockIdx.y * TILE_N;

    // A staging: one float4 per thread (32x32 tile)
    const int ar = tid >> 3;           // 0..31
    const int ac = (tid & 7) * 4;      // 0,4,..,28
    const float* Aptr = A + (size_t)(blockM + ar) * K + ac;

    // B staging: each thread owns a 4(k) x 4(n) sub-block of the 32x128 tile
    const int sbr = (tid >> 5) * 4;    // k base: 0,4,..,28
    const int sbc = (tid & 31) * 4;    // n base: 0,4,..,124
    const float* Bptr = B + blockN + sbc + (size_t)sbr * N;

    float4 aReg = *(const float4*)Aptr;
    float4 bReg[4];
    #pragma unroll
    for (int e = 0; e < 4; ++e)
        bReg[e] = *(const float4*)(Bptr + (size_t)e * N);

    v8f acc0 = {0.f,0.f,0.f,0.f,0.f,0.f,0.f,0.f};
    v8f acc1 = acc0;

    for (int k0 = 0; k0 < K; k0 += TILE_K) {
        // ---- commit staged registers to LDS (fp32 -> bf16) ----
        v4bf av;
        av[0] = (bf16_t)aReg.x; av[1] = (bf16_t)aReg.y;
        av[2] = (bf16_t)aReg.z; av[3] = (bf16_t)aReg.w;
        *(v4bf*)&As[ar * LDSS + ac] = av;              // b64 store

        // 4x4 register transpose of B sub-block -> four b64 stores
        v4bf t0, t1, t2, t3;
        t0[0]=(bf16_t)bReg[0].x; t0[1]=(bf16_t)bReg[1].x; t0[2]=(bf16_t)bReg[2].x; t0[3]=(bf16_t)bReg[3].x;
        t1[0]=(bf16_t)bReg[0].y; t1[1]=(bf16_t)bReg[1].y; t1[2]=(bf16_t)bReg[2].y; t1[3]=(bf16_t)bReg[3].y;
        t2[0]=(bf16_t)bReg[0].z; t2[1]=(bf16_t)bReg[1].z; t2[2]=(bf16_t)bReg[2].z; t2[3]=(bf16_t)bReg[3].z;
        t3[0]=(bf16_t)bReg[0].w; t3[1]=(bf16_t)bReg[1].w; t3[2]=(bf16_t)bReg[2].w; t3[3]=(bf16_t)bReg[3].w;
        *(v4bf*)&BsT[(sbc + 0) * LDSS + sbr] = t0;
        *(v4bf*)&BsT[(sbc + 1) * LDSS + sbr] = t1;
        *(v4bf*)&BsT[(sbc + 2) * LDSS + sbr] = t2;
        *(v4bf*)&BsT[(sbc + 3) * LDSS + sbr] = t3;
        __syncthreads();

        // ---- register prefetch of next K tile (overlaps with WMMA) ----
        if (k0 + TILE_K < K) {
            aReg = *(const float4*)(Aptr + k0 + TILE_K);
            #pragma unroll
            for (int e = 0; e < 4; ++e)
                bReg[e] = *(const float4*)(Bptr + (size_t)(k0 + TILE_K + e) * N);
        }

        // ---- A fragment (contiguous K = hl*16 + s, same as B), two b128 LDS loads ----
        const int am = (waveM * 16 + l16) * LDSS;
        v8bf a0 = *(const v8bf*)&As[am + hl * 16];         // K = hl*16 .. +7
        v8bf a1 = *(const v8bf*)&As[am + hl * 16 + 8];     // K = hl*16+8 .. +15
        v16bf afrag = __builtin_shufflevector(a0, a1,
                0,1,2,3,4,5,6,7,8,9,10,11,12,13,14,15);

        // ---- B fragments: contiguous K in BsT rows ----
        const int bn0 = (waveN * 32 + l16) * LDSS;
        const int bn1 = (waveN * 32 + 16 + l16) * LDSS;
        v8bf b0lo = *(const v8bf*)&BsT[bn0 + hl * 16];
        v8bf b0hi = *(const v8bf*)&BsT[bn0 + hl * 16 + 8];
        v8bf b1lo = *(const v8bf*)&BsT[bn1 + hl * 16];
        v8bf b1hi = *(const v8bf*)&BsT[bn1 + hl * 16 + 8];
        v16bf bfrag0 = __builtin_shufflevector(b0lo, b0hi,
                0,1,2,3,4,5,6,7,8,9,10,11,12,13,14,15);
        v16bf bfrag1 = __builtin_shufflevector(b1lo, b1hi,
                0,1,2,3,4,5,6,7,8,9,10,11,12,13,14,15);

        acc0 = __builtin_amdgcn_wmma_f32_16x16x32_bf16(
                false, afrag, false, bfrag0, (short)0, acc0, false, false);
        acc1 = __builtin_amdgcn_wmma_f32_16x16x32_bf16(
                false, afrag, false, bfrag1, (short)0, acc1, false, false);

        __syncthreads();
    }

    // ---- store C: VGPR j of lane -> row 8*hl + j, col l16 (no guards) ----
    const int grow0 = blockM + waveM * 16 + hl * 8;
    const int gc0   = blockN + waveN * 32 + l16;
    const int gc1   = gc0 + 16;
    const float bv0 = bias ? bias[gc0] : 0.f;
    const float bv1 = bias ? bias[gc1] : 0.f;
    #pragma unroll
    for (int j = 0; j < 8; ++j) {
        float v0 = acc0[j] + bv0;
        float v1 = acc1[j] + bv1;
        if (relu) { v0 = fmaxf(v0, 0.f); v1 = fmaxf(v1, 0.f); }
        C[(size_t)(grow0 + j) * N + gc0] = v0;
        C[(size_t)(grow0 + j) * N + gc1] = v1;
    }
}

// ---------------------------------------------------------------------------
// Simple GEMM for the tiny edge cases (emb: K=16, proj: N=16).
// ---------------------------------------------------------------------------
__global__ __launch_bounds__(256)
void gemm_simple_kernel(const float* __restrict__ A, const float* __restrict__ B,
                        const float* __restrict__ bias, float* __restrict__ C,
                        int M, int N, int K)
{
    const int idx = blockIdx.x * 256 + threadIdx.x;
    if (idx >= M * N) return;
    const int row = idx / N, col = idx % N;
    float s = bias ? bias[col] : 0.f;
    for (int k = 0; k < K; ++k)
        s = fmaf(A[(size_t)row * K + k], B[(size_t)k * N + col], s);
    C[idx] = s;
}

// ---------------------------------------------------------------------------
// Fused attention for one (batch, head): online softmax, K/V staged in LDS
// (async global->LDS copies when available).
// ---------------------------------------------------------------------------
__global__ __launch_bounds__(128)
void attention_kernel(const float* __restrict__ Q, const float* __restrict__ Km,
                      const float* __restrict__ V, float* __restrict__ Out)
{
    const int bh = blockIdx.x;
    const int b  = bh / NHEAD;
    const int h  = bh % NHEAD;
    const int t  = threadIdx.x;

    __shared__ __align__(16) float Ks[SEQ][DHEAD];
    __shared__ __align__(16) float Vs[SEQ][DHEAD];

    #pragma unroll
    for (int i = 0; i < 8; ++i) {            // 128*32 / (128 thr * 4) = 8 float4
        int f = t + i * 128;                  // float4 slot
        int j = f >> 3, d = (f & 7) * 4;
        size_t g = ((size_t)(b * SEQ + j) * D_MODEL) + h * DHEAD + d;
#if HAVE_ASYNC_LDS
        GLOAD_ASYNC_B128(&Km[g], &Ks[j][d]);
        GLOAD_ASYNC_B128(&V[g],  &Vs[j][d]);
#else
        *(float4*)&Ks[j][d] = *(const float4*)&Km[g];
        *(float4*)&Vs[j][d] = *(const float4*)&V[g];
#endif
    }
#if HAVE_ASYNC_LDS
    __builtin_amdgcn_s_wait_asynccnt(0);
#endif
    __syncthreads();

    const float scale = 0.17677669529663687f;   // 1/sqrt(32)
    float qv[DHEAD];
    #pragma unroll
    for (int d = 0; d < DHEAD; d += 4) {
        float4 qq = *(const float4*)&Q[((size_t)(b * SEQ + t) * D_MODEL) + h * DHEAD + d];
        qv[d] = qq.x * scale; qv[d+1] = qq.y * scale;
        qv[d+2] = qq.z * scale; qv[d+3] = qq.w * scale;
    }

    float m = -3.4e38f, l = 0.f;
    float o[DHEAD];
    #pragma unroll
    for (int d = 0; d < DHEAD; ++d) o[d] = 0.f;

    for (int j = 0; j < SEQ; ++j) {
        float s = 0.f;
        #pragma unroll
        for (int d = 0; d < DHEAD; ++d) s = fmaf(qv[d], Ks[j][d], s);
        float nm   = fmaxf(m, s);
        float corr = __expf(m - nm);
        float p    = __expf(s - nm);
        l = l * corr + p;
        #pragma unroll
        for (int d = 0; d < DHEAD; ++d) o[d] = o[d] * corr + p * Vs[j][d];
        m = nm;
    }
    const float inv = 1.f / l;
    #pragma unroll
    for (int d = 0; d < DHEAD; ++d)
        Out[((size_t)(b * SEQ + t) * D_MODEL) + h * DHEAD + d] = o[d] * inv;
}

// ---------------------------------------------------------------------------
// Out = LayerNorm(X + Y) * g + b  over rows of width 256. One wave32 per row.
// ---------------------------------------------------------------------------
__global__ __launch_bounds__(256)
void add_ln_kernel(const float* __restrict__ X, const float* __restrict__ Y,
                   const float* __restrict__ g, const float* __restrict__ bta,
                   float* __restrict__ Out, int M)
{
    const int row  = blockIdx.x * 8 + (threadIdx.x >> 5);
    const int lane = threadIdx.x & 31;
    if (row >= M) return;
    const float* xr = X + (size_t)row * D_MODEL;
    const float* yr = Y + (size_t)row * D_MODEL;

    float v[8], s = 0.f, ss = 0.f;
    #pragma unroll
    for (int i = 0; i < 8; ++i) {
        int c = lane + i * 32;
        float t = xr[c] + yr[c];
        v[i] = t; s += t; ss += t * t;
    }
    #pragma unroll
    for (int off = 16; off >= 1; off >>= 1) {
        s  += __shfl_xor(s,  off, 32);
        ss += __shfl_xor(ss, off, 32);
    }
    const float mean = s * (1.f / D_MODEL);
    const float var  = ss * (1.f / D_MODEL) - mean * mean;
    const float inv  = rsqrtf(var + 1e-5f);
    #pragma unroll
    for (int i = 0; i < 8; ++i) {
        int c = lane + i * 32;
        Out[(size_t)row * D_MODEL + c] = (v[i] - mean) * inv * g[c] + bta[c];
    }
}

// ---------------------------------------------------------------------------
// Fused all-pairs distance + top-KNNS smallest per row + global sum.
// Column tiles streamed through LDS via async copies when available.
// ---------------------------------------------------------------------------
__global__ __launch_bounds__(256)
void dist_topk_kernel(const float* __restrict__ Xr, float* __restrict__ out)
{
    __shared__ __align__(16) float tile[256][D_IN];
    __shared__ float redbuf[8];
    const int tid = threadIdx.x;
    const int row = blockIdx.x * 256 + tid;

    float a[D_IN];
    #pragma unroll
    for (int d = 0; d < D_IN; d += 4)
        *(float4*)&a[d] = *(const float4*)&Xr[(size_t)row * D_IN + d];

    float best[KNNS];
    #pragma unroll
    for (int i = 0; i < KNNS; ++i) best[i] = 3.4e38f;

    for (int jt = 0; jt < NROWS; jt += 256) {
#if HAVE_ASYNC_LDS
        #pragma unroll
        for (int d = 0; d < D_IN; d += 4)
            GLOAD_ASYNC_B128(&Xr[(size_t)(jt + tid) * D_IN + d], &tile[tid][d]);
        __builtin_amdgcn_s_wait_asynccnt(0);
#else
        #pragma unroll
        for (int d = 0; d < D_IN; d += 4)
            *(float4*)&tile[tid][d] = *(const float4*)&Xr[(size_t)(jt + tid) * D_IN + d];
#endif
        __syncthreads();
        #pragma unroll 4
        for (int j = 0; j < 256; ++j) {
            float dist = 0.f;
            #pragma unroll
            for (int d = 0; d < D_IN; d += 4) {
                float4 tv = *(const float4*)&tile[j][d];
                float d0 = a[d]   - tv.x;
                float d1 = a[d+1] - tv.y;
                float d2 = a[d+2] - tv.z;
                float d3 = a[d+3] - tv.w;
                dist = fmaf(d0, d0, dist);
                dist = fmaf(d1, d1, dist);
                dist = fmaf(d2, d2, dist);
                dist = fmaf(d3, d3, dist);
            }
            if (dist < best[KNNS - 1]) {
                float x = dist;                 // branchless sorted insert
                #pragma unroll
                for (int p = 0; p < KNNS; ++p) {
                    float lo = fminf(best[p], x);
                    float hi = fmaxf(best[p], x);
                    best[p] = lo; x = hi;
                }
            }
        }
        __syncthreads();
    }

    float s = 0.f;
    #pragma unroll
    for (int i = 0; i < KNNS; ++i) s += best[i];
    #pragma unroll
    for (int off = 16; off >= 1; off >>= 1) s += __shfl_xor(s, off, 32);
    if ((tid & 31) == 0) redbuf[tid >> 5] = s;
    __syncthreads();
    if (tid == 0) {
        float t = 0.f;
        #pragma unroll
        for (int w = 0; w < 8; ++w) t += redbuf[w];
        atomicAdd(out, t);
    }
}

__global__ void zero_kernel(float* p) { p[0] = 0.f; }

// ---------------------------------------------------------------------------
// Host launcher
// ---------------------------------------------------------------------------
extern "C" void kernel_launch(void* const* d_in, const int* in_sizes, int n_in,
                              void* d_out, int out_size, void* d_ws, size_t ws_size,
                              hipStream_t stream) {
    (void)in_sizes; (void)n_in; (void)out_size; (void)ws_size;

    const float* x     = (const float*)d_in[0];
    const float* W_emb = (const float*)d_in[1];
    const float* b_emb = (const float*)d_in[2];
    const float* Wq    = (const float*)d_in[3];
    const float* Wk    = (const float*)d_in[4];
    const float* Wv    = (const float*)d_in[5];
    const float* Wo    = (const float*)d_in[6];
    const float* ln1_g = (const float*)d_in[7];
    const float* ln1_b = (const float*)d_in[8];
    const float* W1    = (const float*)d_in[9];
    const float* b1    = (const float*)d_in[10];
    const float* W2    = (const float*)d_in[11];
    const float* b2    = (const float*)d_in[12];
    const float* ln2_g = (const float*)d_in[13];
    const float* ln2_b = (const float*)d_in[14];
    const float* Wd    = (const float*)d_in[15];
    const float* bd    = (const float*)d_in[16];
    float* out = (float*)d_out;
    float* ws  = (float*)d_ws;

    // workspace layout (floats); reuse keeps footprint at 56 MB
    const size_t MB2 = 2u * 1024u * 1024u;     // 2M floats = one [8192,256]
    float* h0   = ws + 0 * MB2;
    float* q    = ws + 1 * MB2;
    float* k    = ws + 2 * MB2;
    float* v    = ws + 3 * MB2;
    float* att  = ws + 4 * MB2;
    float* o    = ws + 5 * MB2;
    float* h1   = ws + 6 * MB2;
    float* ff1  = ws + 1 * MB2;                // reuse q..att (8M floats)
    float* ff2  = ws + 5 * MB2;                // reuse o
    float* h2   = ws + 0 * MB2;                // reuse h0
    float* xrec = ws + 1 * MB2;                // reuse ff1 head

    const dim3 blk256(256), blk128(128);
    const dim3 gN256(NROWS / TILE_M, D_MODEL / TILE_N);   // (256, 2)
    const dim3 gN1024(NROWS / TILE_M, D_FF / TILE_N);     // (256, 8)

    // 1) embed (K=16 -> simple kernel)
    gemm_simple_kernel<<<dim3((NROWS * D_MODEL + 255) / 256), blk256, 0, stream>>>(
        x, W_emb, b_emb, h0, NROWS, D_MODEL, D_IN);
    // 2) Q, K, V projections (WMMA)
    gemm_wmma_kernel<<<gN256, blk256, 0, stream>>>(h0, Wq, nullptr, q, NROWS, D_MODEL, D_MODEL, 0);
    gemm_wmma_kernel<<<gN256, blk256, 0, stream>>>(h0, Wk, nullptr, k, NROWS, D_MODEL, D_MODEL, 0);
    gemm_wmma_kernel<<<gN256, blk256, 0, stream>>>(h0, Wv, nullptr, v, NROWS, D_MODEL, D_MODEL, 0);
    // 3) attention (per batch-head, fused online softmax)
    attention_kernel<<<dim3(BATCH * NHEAD), blk128, 0, stream>>>(q, k, v, att);
    // 4) output projection (WMMA)
    gemm_wmma_kernel<<<gN256, blk256, 0, stream>>>(att, Wo, nullptr, o, NROWS, D_MODEL, D_MODEL, 0);
    // 5) residual + LN1
    add_ln_kernel<<<dim3(NROWS / 8), blk256, 0, stream>>>(h0, o, ln1_g, ln1_b, h1, NROWS);
    // 6) FFN (WMMA, ReLU fused into first GEMM)
    gemm_wmma_kernel<<<gN1024, blk256, 0, stream>>>(h1, W1, b1, ff1, NROWS, D_FF, D_MODEL, 1);
    gemm_wmma_kernel<<<gN256, blk256, 0, stream>>>(ff1, W2, b2, ff2, NROWS, D_MODEL, D_FF, 0);
    // 7) residual + LN2
    add_ln_kernel<<<dim3(NROWS / 8), blk256, 0, stream>>>(h1, ff2, ln2_g, ln2_b, h2, NROWS);
    // 8) projection down to D_IN (N=16 -> simple kernel)
    gemm_simple_kernel<<<dim3((NROWS * D_IN + 255) / 256), blk256, 0, stream>>>(
        h2, Wd, bd, xrec, NROWS, D_IN, D_MODEL);
    // 9) fused distance + top-k + sum
    zero_kernel<<<dim3(1), dim3(1), 0, stream>>>(out);
    dist_topk_kernel<<<dim3(NROWS / 256), blk256, 0, stream>>>(xrec, out);
}
